// SinkhornSelfAttention_21328807592574
// MI455X (gfx1250) — matrix operand order repaired and checked by reading, not compiled, not run on hardware
//
#include <hip/hip_runtime.h>
#include <hip/hip_bf16.h>

// Problem constants
#define BB 4
#define TT 4096
#define DD 1024
#define HH 16
#define LHH 4
#define SHH 12
#define DHH 64
#define NB 64   // buckets
#define WW 64   // bucket width

typedef __attribute__((ext_vector_type(16))) _Float16 v16h;
typedef __attribute__((ext_vector_type(4)))  _Float16 v4h;
typedef __attribute__((ext_vector_type(8)))  float    v8f;
typedef __attribute__((ext_vector_type(4)))  unsigned int v4u;
typedef __attribute__((ext_vector_type(4)))  int      v4i;
typedef __attribute__((ext_vector_type(8)))  int      v8i;

#if defined(__has_builtin)
#if __has_builtin(__builtin_amdgcn_tensor_load_to_lds)
#define HAVE_TDM 1
#endif
#endif
#ifndef HAVE_TDM
#define HAVE_TDM 0
#endif

// GEMM LDS tile row stride (f16 elements): 32 k + 8 pad. TDM pad_enable adds
// 4 dwords after every 16 dwords so 16 lanes reading 16 rows hit 16 banks.
#define LDT 40

__device__ __forceinline__ v8f wmma16(v16h a, v16h b, v8f c) {
  return __builtin_amdgcn_wmma_f32_16x16x32_f16(false, a, false, b, (short)0, c,
                                                false, false);
}

// A fragment: logical A[m][k] row-major, leading dim ld. One contiguous
// 16-f16 run per lane -> 2x ds_load_b128.
__device__ __forceinline__ v16h load_a_frag(const _Float16* p, int ld, int lane) {
  int m  = lane & 15;
  int kb = (lane < 16) ? 0 : 16;
  v16h a;
#pragma unroll
  for (int i = 0; i < 16; ++i) {
    int kk = kb + i;
    a[i] = p[m * ld + kk];
  }
  return a;
}

// B fragment: logical B[k][n] row-major (k-major), leading dim ld (strided).
__device__ __forceinline__ v16h load_b_frag(const _Float16* p, int ld, int lane) {
  int n  = lane & 15;
  int kb = (lane < 16) ? 0 : 16;
  v16h b;
#pragma unroll
  for (int i = 0; i < 16; ++i) b[i] = p[(kb + i) * ld + n];
  return b;
}

// B fragment from n-major storage: logical B[k][n] = src[n*ld + k].
// 16 contiguous f16 per lane -> 2x ds_load_b128.
__device__ __forceinline__ v16h load_bt_frag(const _Float16* p, int ld, int lane) {
  int n  = lane & 15;
  int kb = (lane < 16) ? 0 : 16;
  v16h b;
#pragma unroll
  for (int i = 0; i < 16; ++i) b[i] = p[n * ld + kb + i];
  return b;
}

#if HAVE_TDM
// Tensor Data Mover: load a 2D f16 tile (tile_d0 x tile_d1, row stride stride0
// elements in global) into LDS at lds_dst. pad_amt < 0 disables LDS padding.
// Descriptor per CDNA5 ISA 8.3/8.4: count=1, type=2, data_size=1 (2 bytes).
__device__ __forceinline__ void tdm_load_f16(const _Float16* gsrc,
                                             const void* lds_dst,
                                             unsigned tile_d0, unsigned tile_d1,
                                             unsigned long long stride0,
                                             unsigned tensor_d1,
                                             int pad_iv, int pad_amt) {
  unsigned long long ga = (unsigned long long)(uintptr_t)gsrc;
  v4u g0;
  g0.x = 1u;                                    // count=1 (valid), load, user
  g0.y = (unsigned)(uintptr_t)lds_dst;          // low 32 bits = LDS offset
  g0.z = (unsigned)ga;
  g0.w = (unsigned)((ga >> 32) & 0x01FFFFFFu) | (2u << 30);  // type=2 (image)
  unsigned pad_en = (pad_amt >= 0) ? 1u : 0u;
  unsigned w0 = (1u << 16) | (pad_en << 20);
  if (pad_en) w0 |= ((unsigned)pad_iv << 22) | ((unsigned)pad_amt << 25);
  unsigned long long td0 = stride0;             // tensor_dim0 (safe bound)
  v8i g1;
  g1[0] = (int)w0;
  g1[1] = (int)((unsigned)(td0 & 0xFFFFu) << 16);
  g1[2] = (int)((unsigned)((td0 >> 16) & 0xFFFFu) |
                ((tensor_d1 & 0xFFFFu) << 16));
  g1[3] = (int)(((tensor_d1 >> 16) & 0xFFFFu) | (tile_d0 << 16));
  g1[4] = (int)(tile_d1 & 0xFFFFu);             // tile_dim1; tile_dim2=0
  g1[5] = (int)(unsigned)(stride0 & 0xFFFFFFFFu);
  g1[6] = (int)(unsigned)((stride0 >> 32) & 0xFFFFu);
  g1[7] = 0;
  v4i z4 = {0, 0, 0, 0};
#if __clang_major__ >= 23
  v8i z8 = {0, 0, 0, 0, 0, 0, 0, 0};
  __builtin_amdgcn_tensor_load_to_lds(g0, g1, z4, z4, z8, 0);
#else
  __builtin_amdgcn_tensor_load_to_lds(g0, g1, z4, z4, 0);
#endif
}
#endif  // HAVE_TDM

// ---------------------------------------------------------------------------
// f32 -> f16 conversion pass (vectorized x4)
// ---------------------------------------------------------------------------
__global__ __launch_bounds__(256) void cvt_kernel(const float* __restrict__ s,
                                                  _Float16* __restrict__ d,
                                                  int n4) {
  int i = blockIdx.x * 256 + threadIdx.x;
  if (i < n4) {
    float4 v = ((const float4*)s)[i];
    v4h o = {(_Float16)v.x, (_Float16)v.y, (_Float16)v.z, (_Float16)v.w};
    ((v4h*)d)[i] = o;
  }
}

// f32 [K][N] -> f16 transposed [N][K], 32x32 LDS tiles.
__global__ __launch_bounds__(256) void cvt_t_kernel(const float* __restrict__ s,
                                                    _Float16* __restrict__ d,
                                                    int K, int N) {
  __shared__ _Float16 tile[32][33];
  const int kT = K / 32;
  const int k0 = (blockIdx.x % kT) * 32;
  const int n0 = (blockIdx.x / kT) * 32;
  for (int i = threadIdx.x; i < 1024; i += 256) {
    int r = i >> 5, c = i & 31;  // r: k offset, c: n offset
    tile[c][r] = (_Float16)s[(size_t)(k0 + r) * N + n0 + c];
  }
  __syncthreads();
  for (int i = threadIdx.x; i < 1024; i += 256) {
    int r = i >> 5, c = i & 31;  // r: n offset, c: k offset
    d[(size_t)(n0 + r) * K + k0 + c] = tile[r][c];
  }
}

// ---------------------------------------------------------------------------
// 128x128-tile GEMM, f16 in / f32 WMMA accumulate, TDM double-buffered.
// A: [M,K] row-major f16. BT: [N,K] row-major f16 (pre-transposed weights).
// Both tiles are n/m-major with 32 contiguous k -> all frag loads are b128.
// MODE 0: D0 = q (f16 [B,H,T,DH]) ; MODE 1: D0=k, D1=v ; MODE 2: Dout f32+bias
// ---------------------------------------------------------------------------
template <int MODE>
__global__ __launch_bounds__(256) void gemm_kernel(
    const _Float16* __restrict__ Ah, const _Float16* __restrict__ BhT,
    const float* __restrict__ bias, _Float16* __restrict__ D0,
    _Float16* __restrict__ D1, float* __restrict__ Dout, int N, int K) {
  __shared__ _Float16 sA[2][128 * LDT];
  __shared__ _Float16 sB[2][128 * LDT];
  const int tid = threadIdx.x, lane = tid & 31, wave = tid >> 5;
  const int nBlk = N / 128;
  const int row0 = (blockIdx.x / nBlk) * 128;
  const int col0 = (blockIdx.x % nBlk) * 128;
  const int wm = (wave & 3) * 32;   // wave M offset (4 waves over 128 rows)
  const int wn = (wave >> 2) * 64;  // wave N offset (2 waves over 128 cols)
  v8f acc[2][4] = {};
  const int nK = K / 32;

#if HAVE_TDM
  if (tid == 0) {
    tdm_load_f16(Ah + (size_t)row0 * K, &sA[0][0], 32, 128, K, 128, 3, 3);
    tdm_load_f16(BhT + (size_t)col0 * K, &sB[0][0], 32, 128, K, 128, 3, 3);
  }
  for (int it = 0; it < nK; ++it) {
    const int p = it & 1;
    if (tid == 0) __builtin_amdgcn_s_wait_tensorcnt(0);
    __syncthreads();  // current buffers ready; previous reads complete
    if (tid == 0 && it + 1 < nK) {
      tdm_load_f16(Ah + (size_t)row0 * K + (it + 1) * 32, &sA[p ^ 1][0], 32,
                   128, K, 128, 3, 3);
      tdm_load_f16(BhT + (size_t)col0 * K + (it + 1) * 32, &sB[p ^ 1][0], 32,
                   128, K, 128, 3, 3);
    }
    v16h af0 = load_a_frag(&sA[p][wm * LDT], LDT, lane);
    v16h af1 = load_a_frag(&sA[p][(wm + 16) * LDT], LDT, lane);
    v16h bf0 = load_bt_frag(&sB[p][(wn + 0) * LDT], LDT, lane);
    v16h bf1 = load_bt_frag(&sB[p][(wn + 16) * LDT], LDT, lane);
    v16h bf2 = load_bt_frag(&sB[p][(wn + 32) * LDT], LDT, lane);
    v16h bf3 = load_bt_frag(&sB[p][(wn + 48) * LDT], LDT, lane);
    acc[0][0] = wmma16(af0, bf0, acc[0][0]);
    acc[0][1] = wmma16(af0, bf1, acc[0][1]);
    acc[0][2] = wmma16(af0, bf2, acc[0][2]);
    acc[0][3] = wmma16(af0, bf3, acc[0][3]);
    acc[1][0] = wmma16(af1, bf0, acc[1][0]);
    acc[1][1] = wmma16(af1, bf1, acc[1][1]);
    acc[1][2] = wmma16(af1, bf2, acc[1][2]);
    acc[1][3] = wmma16(af1, bf3, acc[1][3]);
  }
#else
  for (int it = 0; it < nK; ++it) {
    const int k0 = it * 32;
    for (int i = tid; i < 128 * 32; i += 256) {
      int r = i >> 5, c = i & 31;
      sA[0][r * LDT + c] = Ah[(size_t)(row0 + r) * K + k0 + c];
      sB[0][r * LDT + c] = BhT[(size_t)(col0 + r) * K + k0 + c];
    }
    __syncthreads();
    v16h af0 = load_a_frag(&sA[0][wm * LDT], LDT, lane);
    v16h af1 = load_a_frag(&sA[0][(wm + 16) * LDT], LDT, lane);
#pragma unroll
    for (int ni = 0; ni < 4; ++ni) {
      v16h bf = load_bt_frag(&sB[0][(wn + ni * 16) * LDT], LDT, lane);
      acc[0][ni] = wmma16(af0, bf, acc[0][ni]);
      acc[1][ni] = wmma16(af1, bf, acc[1][ni]);
    }
    __syncthreads();
  }
#endif

  const int ccol  = lane & 15;
  const int rbase = (lane < 16) ? 0 : 8;
#pragma unroll
  for (int mi = 0; mi < 2; ++mi)
#pragma unroll
    for (int ni = 0; ni < 4; ++ni)
#pragma unroll
      for (int r = 0; r < 8; ++r) {
        int m = row0 + wm + mi * 16 + rbase + r;
        int c = col0 + wn + ni * 16 + ccol;
        float val = acc[mi][ni][r];
        if (MODE == 0) {
          int b = m >> 12, t = m & (TT - 1);
          int hh = c >> 6, d = c & 63;
          D0[((size_t)(b * HH + hh) * TT + t) * DHH + d] = (_Float16)val;
        } else if (MODE == 1) {
          int b = m >> 12, t = m & (TT - 1);
          _Float16* dst = (c < 1024) ? D0 : D1;
          int cc = c & 1023;
          int hh = cc >> 6, d = cc & 63;
          dst[((size_t)(b * HH + hh) * TT + t) * DHH + d] = (_Float16)val;
        } else {
          Dout[(size_t)m * DD + c] = val + bias[c];
        }
      }
}

// ---------------------------------------------------------------------------
// Local look-around attention: one WG per (b, h<LH, bucket n).
// dots = Q(64x64) @ Kwin(192x64)^T (no scale), mask pads, softmax, P @ Vwin.
// ---------------------------------------------------------------------------
__global__ __launch_bounds__(256) void local_attn_kernel(
    const _Float16* __restrict__ qh, const _Float16* __restrict__ kh,
    const _Float16* __restrict__ vh, _Float16* __restrict__ aout) {
  __shared__ _Float16 sQ[64][64];
  __shared__ _Float16 sK[192][64];
  __shared__ _Float16 sV[192][64];
  __shared__ float    sD[64][192];
  __shared__ _Float16 sP[64][192];
  const int blk = blockIdx.x;  // B*LH*NB
  const int b = blk >> 8;
  const int h = (blk >> 6) & 3;
  const int n = blk & 63;
  const int tid = threadIdx.x, lane = tid & 31, wave = tid >> 5;
  const size_t base = (size_t)(b * HH + h) * TT * DHH;
  const int j0 = (n == 0) ? 64 : 0;
  const int j1 = (n == NB - 1) ? 128 : 192;

#if HAVE_TDM
  if (tid == 0) {
    tdm_load_f16(qh + base + (size_t)n * WW * DHH, &sQ[0][0], 64, 64, 64, 64,
                 -1, -1);
    tdm_load_f16(kh + base + (size_t)((n - 1) * 64 + j0) * DHH, &sK[j0][0], 64,
                 (unsigned)(j1 - j0), 64, (unsigned)(j1 - j0), -1, -1);
    tdm_load_f16(vh + base + (size_t)((n - 1) * 64 + j0) * DHH, &sV[j0][0], 64,
                 (unsigned)(j1 - j0), 64, (unsigned)(j1 - j0), -1, -1);
    __builtin_amdgcn_s_wait_tensorcnt(0);
  }
  if (n == 0 || n == NB - 1) {  // fill masked-out pad rows (finite values)
    int jb = (n == 0) ? 0 : 128;
    for (int i = tid; i < 64 * 64; i += 256) {
      int j = jb + (i >> 6), d = i & 63;
      sK[j][d] = (_Float16)(-1.0f);
      sV[j][d] = (_Float16)(-1.0f);
    }
  }
  __syncthreads();
#else
  for (int i = tid; i < 64 * 64; i += 256) {
    int t = i >> 6, d = i & 63;
    sQ[t][d] = qh[base + (size_t)(n * 64 + t) * DHH + d];
  }
  for (int i = tid; i < 192 * 64; i += 256) {
    int j = i >> 6, d = i & 63;
    int tj = (n - 1) * 64 + j;
    bool valid = (tj >= 0) && (tj < TT);
    sK[j][d] = valid ? kh[base + (size_t)tj * DHH + d] : (_Float16)(-1.0f);
    sV[j][d] = valid ? vh[base + (size_t)tj * DHH + d] : (_Float16)(-1.0f);
  }
  __syncthreads();
#endif

  for (int tile = wave; tile < 48; tile += 8) {  // 4(M) x 12(N), K=64
    int m0 = (tile / 12) * 16;
    int n0 = (tile % 12) * 16;
    v8f acc = {};
#pragma unroll
    for (int k0 = 0; k0 < 64; k0 += 32) {
      v16h a = load_a_frag(&sQ[m0][k0], 64, lane);
      v16h bb = load_bt_frag(&sK[n0][k0], 64, lane);
      acc = wmma16(a, bb, acc);
    }
    int rb = (lane < 16) ? 0 : 8;
    int cc = lane & 15;
#pragma unroll
    for (int r = 0; r < 8; ++r) sD[m0 + rb + r][n0 + cc] = acc[r];
  }
  __syncthreads();

  if (tid < 64) {
    int r = tid;
    float mx = -3.402823466e38f;
    for (int j = j0; j < j1; ++j) mx = fmaxf(mx, sD[r][j]);
    float sum = 0.0f;
    for (int j = j0; j < j1; ++j) {
      float e = __expf(sD[r][j] - mx);
      sD[r][j] = e;
      sum += e;
    }
    float inv = 1.0f / sum;
    for (int j = 0; j < 192; ++j)
      sP[r][j] = (j >= j0 && j < j1) ? (_Float16)(sD[r][j] * inv)
                                     : (_Float16)0.0f;
  }
  __syncthreads();

  for (int tile = wave; tile < 16; tile += 8) {  // 4x4 tiles, K=192
    int m0 = (tile >> 2) * 16;
    int n0 = (tile & 3) * 16;
    v8f acc = {};
#pragma unroll
    for (int k0 = 0; k0 < 192; k0 += 32) {
      v16h a = load_a_frag(&sP[m0][k0], 192, lane);
      v16h bb = load_b_frag(&sV[k0][n0], 64, lane);
      acc = wmma16(a, bb, acc);
    }
    int rb = (lane < 16) ? 0 : 8;
    int cc = lane & 15;
#pragma unroll
    for (int r = 0; r < 8; ++r) {
      int t = n * 64 + m0 + rb + r;
      aout[((size_t)b * TT + t) * (HH * DHH) + h * DHH + n0 + cc] =
          (_Float16)acc[r];
    }
  }
}

// ---------------------------------------------------------------------------
// Sink sort: per (b,s): feats = [sum_t bq, sum_t bk]; R = softmax(relu(feats@W))
// ---------------------------------------------------------------------------
__global__ __launch_bounds__(256) void sink_sort_kernel(
    const _Float16* __restrict__ qh, const _Float16* __restrict__ kh,
    const float* __restrict__ sortw, _Float16* __restrict__ Rb) {
  __shared__ float sF[64][128];
  __shared__ float sR[64][64];
  const int blk = blockIdx.x;  // 48
  const int b = blk / SHH, s = blk % SHH, h = LHH + s;
  const size_t base = (size_t)(b * HH + h) * TT * DHH;
  const int tid = threadIdx.x;

  for (int i = tid; i < 64 * 128; i += 256) {
    int u = i >> 7, c = i & 127;
    int d = c & 63;
    const _Float16* src = ((c < 64) ? qh : kh) + base + (size_t)u * WW * DHH + d;
    float sum = 0.0f;
    for (int t = 0; t < WW; ++t) sum += (float)src[t * DHH];
    sF[u][c] = sum;
  }
  __syncthreads();
  for (int i = tid; i < 64 * 64; i += 256) {
    int u = i >> 6, v = i & 63;
    float sum = 0.0f;
    for (int c = 0; c < 128; ++c)
      sum += sF[u][c] * sortw[((size_t)s * 128 + c) * NB + v];
    sR[u][v] = fmaxf(sum, 0.0f);
  }
  __syncthreads();
  if (tid < 64) {
    int u = tid;
    float mx = -3.402823466e38f;
    for (int v = 0; v < 64; ++v) mx = fmaxf(mx, sR[u][v]);
    float sum = 0.0f;
    for (int v = 0; v < 64; ++v) {
      float e = __expf(sR[u][v] - mx);
      sR[u][v] = e;
      sum += e;
    }
    float inv = 1.0f / sum;
    for (int v = 0; v < 64; ++v)
      Rb[(size_t)blk * 4096 + u * 64 + v] = (_Float16)(sR[u][v] * inv);
  }
}

// ---------------------------------------------------------------------------
// Sink mix: bk_r = R @ bk, bv_r = R @ bv (M=64, K=64, N=4096 in 128 chunks)
// ---------------------------------------------------------------------------
__global__ __launch_bounds__(256) void sink_mix_kernel(
    const _Float16* __restrict__ Rb, const _Float16* __restrict__ kh,
    const _Float16* __restrict__ vh, _Float16* __restrict__ bkr,
    _Float16* __restrict__ bvr) {
  __shared__ _Float16 sA[64][64];
  __shared__ _Float16 sB2[64][128];
  int idx = blockIdx.x;  // 2*48*32
  const int kv = idx / 1536;
  idx %= 1536;
  const int bs = idx / 32;
  const int chunk = idx % 32;
  const int b = bs / SHH, s = bs % SHH, h = LHH + s;
  const size_t base = (size_t)(b * HH + h) * TT * DHH;
  const int c0 = chunk * 128;
  const _Float16* src = kv ? vh : kh;
  _Float16* dst = kv ? bvr : bkr;
  const int tid = threadIdx.x, lane = tid & 31, wave = tid >> 5;

#if HAVE_TDM
  if (tid == 0) {
    tdm_load_f16(Rb + (size_t)bs * 4096, &sA[0][0], 64, 64, 64, 64, -1, -1);
    tdm_load_f16(src + base + c0, &sB2[0][0], 128, 64, 4096, 64, -1, -1);
    __builtin_amdgcn_s_wait_tensorcnt(0);
  }
  __syncthreads();
#else
  for (int i = tid; i < 64 * 64; i += 256)
    sA[i >> 6][i & 63] = Rb[(size_t)bs * 4096 + i];
  for (int i = tid; i < 64 * 128; i += 256) {
    int vr = i >> 7, cc = i & 127;
    sB2[vr][cc] = src[base + (size_t)vr * 4096 + c0 + cc];
  }
  __syncthreads();
#endif

  for (int tile = wave * 4; tile < wave * 4 + 4; ++tile) {  // 4x8 tiles
    int m0 = (tile / 8) * 16;
    int n0 = (tile % 8) * 16;
    v8f acc = {};
#pragma unroll
    for (int k0 = 0; k0 < 64; k0 += 32) {
      v16h a = load_a_frag(&sA[m0][k0], 64, lane);
      v16h bb = load_b_frag(&sB2[k0][n0], 128, lane);
      acc = wmma16(a, bb, acc);
    }
    int rb = (lane < 16) ? 0 : 8;
    int cc = lane & 15;
#pragma unroll
    for (int r = 0; r < 8; ++r) {
      int u = m0 + rb + r;
      dst[(size_t)bs * 262144 + (size_t)u * 4096 + c0 + n0 + cc] =
          (_Float16)acc[r];
    }
  }
}

// ---------------------------------------------------------------------------
// Sink attention: per (b,s,u): softmax(q @ [bk_r;bk]^T * dh^-0.5) @ [bv_r;bv]
// ---------------------------------------------------------------------------
__global__ __launch_bounds__(256) void sink_attn_kernel(
    const _Float16* __restrict__ qh, const _Float16* __restrict__ kh,
    const _Float16* __restrict__ vh, const _Float16* __restrict__ bkr,
    const _Float16* __restrict__ bvr, _Float16* __restrict__ aout) {
  __shared__ _Float16 sQ[64][64];
  __shared__ _Float16 sK[128][64];
  __shared__ _Float16 sV[128][64];
  __shared__ float    sD[64][128];
  __shared__ _Float16 sP[64][128];
  const int blk = blockIdx.x;  // 48*64
  const int bs = blk >> 6, u = blk & 63;
  const int b = bs / SHH, s = bs % SHH, h = LHH + s;
  const size_t base = (size_t)(b * HH + h) * TT * DHH;
  const int tid = threadIdx.x, lane = tid & 31, wave = tid >> 5;

#if HAVE_TDM
  if (tid == 0) {
    tdm_load_f16(qh + base + (size_t)u * 4096, &sQ[0][0], 64, 64, 64, 64, -1, -1);
    tdm_load_f16(bkr + (size_t)bs * 262144 + (size_t)u * 4096, &sK[0][0], 64,
                 64, 64, 64, -1, -1);
    tdm_load_f16(kh + base + (size_t)u * 4096, &sK[64][0], 64, 64, 64, 64, -1, -1);
    tdm_load_f16(bvr + (size_t)bs * 262144 + (size_t)u * 4096, &sV[0][0], 64,
                 64, 64, 64, -1, -1);
    tdm_load_f16(vh + base + (size_t)u * 4096, &sV[64][0], 64, 64, 64, 64, -1, -1);
    __builtin_amdgcn_s_wait_tensorcnt(0);
  }
  __syncthreads();
#else
  for (int i = tid; i < 64 * 64; i += 256) {
    int t = i >> 6, d = i & 63;
    sQ[t][d] = qh[base + (size_t)(u * 64 + t) * DHH + d];
  }
  for (int i = tid; i < 128 * 64; i += 256) {
    int j = i >> 6, d = i & 63;
    if (j < 64) {
      sK[j][d] = bkr[(size_t)bs * 262144 + (size_t)u * 4096 + j * 64 + d];
      sV[j][d] = bvr[(size_t)bs * 262144 + (size_t)u * 4096 + j * 64 + d];
    } else {
      sK[j][d] = kh[base + (size_t)(u * 64 + (j - 64)) * DHH + d];
      sV[j][d] = vh[base + (size_t)(u * 64 + (j - 64)) * DHH + d];
    }
  }
  __syncthreads();
#endif

  for (int tile = wave * 4; tile < wave * 4 + 4; ++tile) {  // 4x8 tiles, K=64
    int m0 = (tile / 8) * 16;
    int n0 = (tile % 8) * 16;
    v8f acc = {};
#pragma unroll
    for (int k0 = 0; k0 < 64; k0 += 32) {
      v16h a = load_a_frag(&sQ[m0][k0], 64, lane);
      v16h bb = load_bt_frag(&sK[n0][k0], 64, lane);
      acc = wmma16(a, bb, acc);
    }
    int rb = (lane < 16) ? 0 : 8;
    int cc = lane & 15;
#pragma unroll
    for (int r = 0; r < 8; ++r) sD[m0 + rb + r][n0 + cc] = acc[r];
  }
  __syncthreads();

  if (tid < 64) {
    int r = tid;
    const float scale = 0.125f;  // dh^-0.5
    float mx = -3.402823466e38f;
    for (int j = 0; j < 128; ++j) mx = fmaxf(mx, sD[r][j] * scale);
    float sum = 0.0f;
    for (int j = 0; j < 128; ++j) {
      float e = __expf(sD[r][j] * scale - mx);
      sD[r][j] = e;
      sum += e;
    }
    float inv = 1.0f / sum;
    for (int j = 0; j < 128; ++j) sP[r][j] = (_Float16)(sD[r][j] * inv);
  }
  __syncthreads();

  for (int tile = wave; tile < 16; tile += 8) {  // 4x4 tiles, K=128
    int m0 = (tile >> 2) * 16;
    int n0 = (tile & 3) * 16;
    v8f acc = {};
#pragma unroll
    for (int k0 = 0; k0 < 128; k0 += 32) {
      v16h a = load_a_frag(&sP[m0][k0], 128, lane);
      v16h bb = load_b_frag(&sV[k0][n0], 64, lane);
      acc = wmma16(a, bb, acc);
    }
    int rb = (lane < 16) ? 0 : 8;
    int cc = lane & 15;
#pragma unroll
    for (int r = 0; r < 8; ++r) {
      int t = u * 64 + m0 + rb + r;
      aout[((size_t)b * TT + t) * (HH * DHH) + h * DHH + n0 + cc] =
          (_Float16)acc[r];
    }
  }
}

// ---------------------------------------------------------------------------
extern "C" void kernel_launch(void* const* d_in, const int* in_sizes, int n_in,
                              void* d_out, int out_size, void* d_ws,
                              size_t ws_size, hipStream_t stream) {
  (void)in_sizes; (void)n_in; (void)out_size; (void)ws_size;
  const float* x     = (const float*)d_in[0];
  const float* w_q   = (const float*)d_in[1];
  const float* w_kv  = (const float*)d_in[2];
  const float* w_out = (const float*)d_in[3];
  const float* b_out = (const float*)d_in[4];
  const float* sortw = (const float*)d_in[5];
  float* out = (float*)d_out;

  char* ws = (char*)d_ws;
  size_t off = 0;
  const size_t xB = (size_t)16384 * 1024 * 2;  // x / qkv-sized f16 buffers
  _Float16* xh   = (_Float16*)(ws + off); off += xB;  // f16(x) [M,K]
  _Float16* wqT  = (_Float16*)(ws + off); off += (size_t)1024 * 1024 * 2;
  _Float16* wkvT = (_Float16*)(ws + off); off += (size_t)1024 * 2048 * 2;
  _Float16* woT  = (_Float16*)(ws + off); off += (size_t)1024 * 1024 * 2;
  _Float16* qh   = (_Float16*)(ws + off); off += xB;
  _Float16* kh   = (_Float16*)(ws + off); off += xB;
  _Float16* vh   = (_Float16*)(ws + off); off += xB;
  _Float16* ah   = (_Float16*)(ws + off); off += xB;  // pre-out-proj [M,1024]
  _Float16* Rb   = (_Float16*)(ws + off); off += (size_t)48 * 4096 * 2;
  _Float16* bkr  = (_Float16*)(ws + off); off += (size_t)48 * 262144 * 2;
  _Float16* bvr  = (_Float16*)(ws + off);

  dim3 blk(256);
  cvt_kernel<<<(16384 * 1024 / 4) / 256, blk, 0, stream>>>(x, xh, 16384 * 256);
  // weights: f32 [K,N] -> f16 transposed [N,K]
  cvt_t_kernel<<<(1024 / 32) * (1024 / 32), blk, 0, stream>>>(w_q, wqT, 1024, 1024);
  cvt_t_kernel<<<(1024 / 32) * (2048 / 32), blk, 0, stream>>>(w_kv, wkvT, 1024, 2048);
  cvt_t_kernel<<<(1024 / 32) * (1024 / 32), blk, 0, stream>>>(w_out, woT, 1024, 1024);

  gemm_kernel<0><<<(16384 / 128) * (1024 / 128), blk, 0, stream>>>(
      xh, wqT, nullptr, qh, nullptr, nullptr, 1024, 1024);
  gemm_kernel<1><<<(16384 / 128) * (2048 / 128), blk, 0, stream>>>(
      xh, wkvT, nullptr, kh, vh, nullptr, 2048, 1024);
  local_attn_kernel<<<BB * LHH * NB, blk, 0, stream>>>(qh, kh, vh, ah);
  sink_sort_kernel<<<BB * SHH, blk, 0, stream>>>(qh, kh, sortw, Rb);
  sink_mix_kernel<<<2 * 48 * 32, blk, 0, stream>>>(Rb, kh, vh, bkr, bvr);
  sink_attn_kernel<<<48 * 64, blk, 0, stream>>>(qh, kh, vh, bkr, bvr, ah);
  gemm_kernel<2><<<(16384 / 128) * (1024 / 128), blk, 0, stream>>>(
      ah, woT, b_out, nullptr, nullptr, out, 1024, 1024);
}
